// MultiHeadAttention_5617817223706
// MI455X (gfx1250) — matrix operand order repaired and checked by reading, not compiled, not run on hardware
//
#include <hip/hip_runtime.h>
#include <hip/hip_bf16.h>

// Problem constants (match reference)
#define BB 2
#define SS 2048
#define DD 2048
#define HH 16
#define HDD 128
#define MT (BB*SS)            // 4096 rows for all GEMMs
#define BSD ((size_t)BB*SS*DD) // 8388608
#define DSQ ((size_t)DD*DD)    // 4194304

typedef __attribute__((ext_vector_type(16))) __bf16 v16bf;
typedef __attribute__((ext_vector_type(8)))  float  v8f;

union Frag {
    v16bf v;
    unsigned int u[8];
    uint4 q[2];
};

__device__ inline unsigned short f32_to_bf16(float f) {
    unsigned int x = __float_as_uint(f);
    unsigned int r = (x + 0x7FFFu + ((x >> 16) & 1u)) >> 16;
    return (unsigned short)r;
}

__device__ inline int koffA(int v, int g) {
    // 16-bit A-matrix 16x32 layout (contiguous, matches B): vgpr v holds K pair 16g+2v
    return (v << 1) + (g << 4);
}

// A-fragment: rowp = &A[row*ldk + kbase], g = lane>>4
__device__ inline void load_fragA(Frag& f, const unsigned short* rowp, int g) {
#pragma unroll
    for (int v = 0; v < 8; ++v)
        f.u[v] = *reinterpret_cast<const unsigned int*>(rowp + koffA(v, g));
}

// B-fragment: p = &B[col*ldk + kbase + g*16]; 16 contiguous bf16 (32 bytes)
__device__ inline void load_fragB(Frag& f, const unsigned short* p) {
    f.q[0] = *reinterpret_cast<const uint4*>(p);
    f.q[1] = *reinterpret_cast<const uint4*>(p + 8);
}

__device__ inline v8f wmma_bf16(const Frag& a, const Frag& b, v8f c) {
    return __builtin_amdgcn_wmma_f32_16x16x32_bf16(
        false, a.v, false, b.v, (short)0, c, false, false);
}

// ---------------------------------------------------------------- convert
__global__ void cvt_f32_bf16(const float* __restrict__ in,
                             unsigned short* __restrict__ out, int n) {
    int i = blockIdx.x * blockDim.x + threadIdx.x;
    int stride = gridDim.x * blockDim.x;
    for (; i < n; i += stride) out[i] = f32_to_bf16(in[i]);
}

// ---------------------------------------------------------------- GEMM
// out[m,n] = sum_k A[m,k]*W[n,k] + bias[n]     (A: MTxD bf16, W: DxD bf16)
// mode 0: Q  -> bf16 permuted (B,H,S,HD)
// mode 1: K  -> bf16 permuted + f32 permuted (d_out section)
// mode 2: V  -> bf16 transposed (B,H,HD,S) + f32 permuted (d_out section)
// mode 3: O  -> f32 row-major (B,S,D) only
__global__ __launch_bounds__(256)
void wmma_gemm_bf16(const unsigned short* __restrict__ A,
                    const unsigned short* __restrict__ W,
                    const float* __restrict__ bias,
                    unsigned short* __restrict__ out_bf16,
                    float* __restrict__ out_f32,
                    int mode) {
    const int lane = threadIdx.x & 31;
    const int wid  = threadIdx.x >> 5;
    const int wm = wid >> 2, wn = wid & 3;
    const int g = lane >> 4, ln = lane & 15;
    const long moff = (long)blockIdx.y * 64 + (long)wm * 32;
    const long noff = (long)blockIdx.x * 256 + (long)wn * 64;

    v8f acc[2][4];
#pragma unroll
    for (int mt = 0; mt < 2; ++mt)
#pragma unroll
        for (int nt = 0; nt < 4; ++nt)
            acc[mt][nt] = (v8f){0.f,0.f,0.f,0.f,0.f,0.f,0.f,0.f};

    for (int k0 = 0; k0 < DD; k0 += 32) {
        Frag a[2], b[4];
#pragma unroll
        for (int mt = 0; mt < 2; ++mt)
            load_fragA(a[mt], A + (size_t)(moff + mt * 16 + ln) * DD + k0, g);
#pragma unroll
        for (int nt = 0; nt < 4; ++nt)
            load_fragB(b[nt], W + (size_t)(noff + nt * 16 + ln) * DD + k0 + (g << 4));
#pragma unroll
        for (int mt = 0; mt < 2; ++mt)
#pragma unroll
            for (int nt = 0; nt < 4; ++nt)
                acc[mt][nt] = wmma_bf16(a[mt], b[nt], acc[mt][nt]);
    }

    float bvv[4];
#pragma unroll
    for (int nt = 0; nt < 4; ++nt) bvv[nt] = bias[noff + nt * 16 + ln];

#pragma unroll
    for (int mt = 0; mt < 2; ++mt) {
#pragma unroll
        for (int nt = 0; nt < 4; ++nt) {
#pragma unroll
            for (int v = 0; v < 8; ++v) {
                float val = acc[mt][nt][v] + bvv[nt];
                long m = moff + mt * 16 + v + 8 * g;
                long n = noff + nt * 16 + ln;
                if (mode == 3) {
                    out_f32[(size_t)m * DD + n] = val;
                } else {
                    int bz = (int)(m >> 11), s = (int)(m & 2047);
                    int h  = (int)(n >> 7),  hd = (int)(n & 127);
                    size_t pidx = ((((size_t)bz * HH + h) * SS) + s) * HDD + hd;
                    if (mode == 2) {
                        size_t vidx = (((size_t)bz * HH + h) * HDD + hd) * (size_t)SS + s;
                        out_bf16[vidx] = f32_to_bf16(val);
                        out_f32[pidx] = val;
                    } else {
                        out_bf16[pidx] = f32_to_bf16(val);
                        if (mode == 1) out_f32[pidx] = val;
                    }
                }
            }
        }
    }
}

// ---------------------------------------------------------------- flash attention
// Q,K bf16 in (B,H,S,HD); V bf16 transposed (B,H,HD,S); ctx bf16 (B,S,D)
#define FWAVES 8
__global__ __launch_bounds__(256)
void flash_attn_bf16(const unsigned short* __restrict__ Qw,
                     const unsigned short* __restrict__ Kw,
                     const unsigned short* __restrict__ Vt,
                     unsigned short* __restrict__ Ctx) {
    __shared__ unsigned short pbuf[FWAVES * 16 * 36];

    const int lane = threadIdx.x & 31;
    const int wid  = threadIdx.x >> 5;
    const int g = lane >> 4, ln = lane & 15;
    const int h = blockIdx.y, bz = blockIdx.z;
    const int qbase = (blockIdx.x * FWAVES + wid) * 16;

    const size_t head = ((size_t)bz * HH + h) * (size_t)SS * HDD;
    const unsigned short* Qp = Qw + head;
    const unsigned short* Kp = Kw + head;
    const unsigned short* Vp = Vt + head;   // (HD,S) within head
    unsigned short* pb = pbuf + wid * 16 * 36;

    Frag qa[4];
#pragma unroll
    for (int c = 0; c < 4; ++c)
        load_fragA(qa[c], Qp + (size_t)(qbase + ln) * HDD + c * 32, g);

    v8f acc[8];
#pragma unroll
    for (int n = 0; n < 8; ++n) acc[n] = (v8f){0.f,0.f,0.f,0.f,0.f,0.f,0.f,0.f};
    float m_i[8], l_i[8];
#pragma unroll
    for (int v = 0; v < 8; ++v) { m_i[v] = -1.0e30f; l_i[v] = 0.0f; }

    const float sc = 0.08838834764831845f; // 1/sqrt(128)
    const int mrow = qbase + g * 8;
    const int nck = (qbase + 15) / 32 + 1;

    for (int ck = 0; ck < nck; ++ck) {
        const int k0 = ck * 32;
        v8f s0 = (v8f){0.f,0.f,0.f,0.f,0.f,0.f,0.f,0.f};
        v8f s1 = s0;
#pragma unroll
        for (int c = 0; c < 4; ++c) {
            Frag kb;
            load_fragB(kb, Kp + (size_t)(k0 + ln) * HDD + c * 32 + (g << 4));
            s0 = wmma_bf16(qa[c], kb, s0);
        }
#pragma unroll
        for (int c = 0; c < 4; ++c) {
            Frag kb;
            load_fragB(kb, Kp + (size_t)(k0 + 16 + ln) * HDD + c * 32 + (g << 4));
            s1 = wmma_bf16(qa[c], kb, s1);
        }

        // causal mask + scale + online softmax stats
        const int n0 = k0 + ln, n1 = n0 + 16;
        float e0[8], e1[8], mx[8];
#pragma unroll
        for (int v = 0; v < 8; ++v) {
            int m = mrow + v;
            e0[v] = (n0 <= m) ? s0[v] * sc : -1.0e30f;
            e1[v] = (n1 <= m) ? s1[v] * sc : -1.0e30f;
            mx[v] = fmaxf(e0[v], e1[v]);
        }
#pragma unroll
        for (int v = 0; v < 8; ++v) {
#pragma unroll
            for (int d = 1; d < 16; d <<= 1)
                mx[v] = fmaxf(mx[v], __shfl_xor(mx[v], d));
        }
        float p0[8], p1[8], rs[8], al[8];
#pragma unroll
        for (int v = 0; v < 8; ++v) {
            float nm = fmaxf(m_i[v], mx[v]);
            al[v] = __expf(m_i[v] - nm);
            m_i[v] = nm;
            p0[v] = __expf(e0[v] - nm);
            p1[v] = __expf(e1[v] - nm);
            rs[v] = p0[v] + p1[v];
        }
#pragma unroll
        for (int v = 0; v < 8; ++v) {
#pragma unroll
            for (int d = 1; d < 16; d <<= 1)
                rs[v] += __shfl_xor(rs[v], d);
            l_i[v] = l_i[v] * al[v] + rs[v];
        }
#pragma unroll
        for (int n = 0; n < 8; ++n)
#pragma unroll
            for (int v = 0; v < 8; ++v)
                acc[n][v] *= al[v];

        // transpose P (C-layout -> A-layout) through LDS as bf16
        asm volatile("" ::: "memory");
#pragma unroll
        for (int v = 0; v < 8; ++v) {
            int ml = v + 8 * g;
            pb[ml * 36 + ln]      = f32_to_bf16(p0[v]);
            pb[ml * 36 + 16 + ln] = f32_to_bf16(p1[v]);
        }
        asm volatile("" ::: "memory");
        Frag pa;
#pragma unroll
        for (int v = 0; v < 8; ++v)
            pa.u[v] = *reinterpret_cast<const unsigned int*>(pb + ln * 36 + koffA(v, g));
        asm volatile("" ::: "memory");

        // ctx += P @ V   (V transposed: column of hd contiguous over keys)
#pragma unroll
        for (int n = 0; n < 8; ++n) {
            Frag vb;
            load_fragB(vb, Vp + (size_t)(n * 16 + ln) * SS + k0 + (g << 4));
            acc[n] = wmma_bf16(pa, vb, acc[n]);
        }
    }

    // normalize and store ctx (B,S,D) bf16
#pragma unroll
    for (int v = 0; v < 8; ++v) {
        float inv = 1.0f / l_i[v];
        int sq = qbase + v + 8 * g;
#pragma unroll
        for (int n = 0; n < 8; ++n) {
            float val = acc[n][v] * inv;
            size_t idx = ((size_t)bz * SS + sq) * (size_t)DD + h * HDD + n * 16 + ln;
            Ctx[idx] = f32_to_bf16(val);
        }
    }
}

// ---------------------------------------------------------------- launch
extern "C" void kernel_launch(void* const* d_in, const int* in_sizes, int n_in,
                              void* d_out, int out_size, void* d_ws, size_t ws_size,
                              hipStream_t stream) {
    const float* hs = (const float*)d_in[0];
    // d_in[1] = attn_mask (pure causal; applied analytically)
    const float* Wq = (const float*)d_in[2]; const float* bq = (const float*)d_in[3];
    const float* Wk = (const float*)d_in[4]; const float* bk = (const float*)d_in[5];
    const float* Wv = (const float*)d_in[6]; const float* bv = (const float*)d_in[7];
    const float* Wo = (const float*)d_in[8]; const float* bo = (const float*)d_in[9];

    float* out   = (float*)d_out;            // (B,S,D)
    float* k_out = out + BSD;                // (B,H,S,HD)
    float* v_out = out + 2 * BSD;            // (B,H,S,HD)

    unsigned short* ws    = (unsigned short*)d_ws;
    unsigned short* hs_bf = ws;              // B*S*D
    unsigned short* wq_bf = hs_bf + BSD;     // D*D
    unsigned short* wk_bf = wq_bf + DSQ;
    unsigned short* wv_bf = wk_bf + DSQ;
    unsigned short* wo_bf = wv_bf + DSQ;
    unsigned short* q_bf  = wo_bf + DSQ;     // (B,H,S,HD)
    unsigned short* k_bf  = q_bf + BSD;      // (B,H,S,HD)
    unsigned short* vt_bf = k_bf + BSD;      // (B,H,HD,S)
    unsigned short* ctxbf = vt_bf + BSD;     // (B,S,D)

    cvt_f32_bf16<<<2048, 256, 0, stream>>>(hs, hs_bf, (int)BSD);
    cvt_f32_bf16<<<1024, 256, 0, stream>>>(Wq, wq_bf, (int)DSQ);
    cvt_f32_bf16<<<1024, 256, 0, stream>>>(Wk, wk_bf, (int)DSQ);
    cvt_f32_bf16<<<1024, 256, 0, stream>>>(Wv, wv_bf, (int)DSQ);
    cvt_f32_bf16<<<1024, 256, 0, stream>>>(Wo, wo_bf, (int)DSQ);

    dim3 ggrid(DD / 256, MT / 64);           // (8, 64)
    wmma_gemm_bf16<<<ggrid, 256, 0, stream>>>(hs_bf, wq_bf, bq, q_bf, nullptr, 0);
    wmma_gemm_bf16<<<ggrid, 256, 0, stream>>>(hs_bf, wk_bf, bk, k_bf, k_out, 1);
    wmma_gemm_bf16<<<ggrid, 256, 0, stream>>>(hs_bf, wv_bf, bv, vt_bf, v_out, 2);

    dim3 fgrid(SS / (16 * FWAVES), HH, BB);  // (16, 16, 2)
    flash_attn_bf16<<<fgrid, 256, 0, stream>>>(q_bf, k_bf, vt_bf, ctxbf);

    wmma_gemm_bf16<<<ggrid, 256, 0, stream>>>(ctxbf, wo_bf, bo, nullptr, out, 3);
}
